// Attention_20598663151671
// MI455X (gfx1250) — matrix operand order repaired and checked by reading, not compiled, not run on hardware
//
#include <hip/hip_runtime.h>
#include <hip/hip_bf16.h>
#include <stdint.h>

// ---------------------------------------------------------------------------
// Attention block for MI455X (gfx1250).
//   bf16 WMMA for all GEMMs, 64-wide flash attention tiles, CDNA5
//   async-to-LDS staging (double buffered), exp2-domain online softmax,
//   batched fragment loads so WMMAs issue back-to-back.
// ---------------------------------------------------------------------------

typedef __bf16 bf16_t;
typedef __attribute__((ext_vector_type(16))) __bf16 v16bf;
typedef __attribute__((ext_vector_type(8)))  float  v8f;

union Frag16 { v16bf v; uint32_t u[8]; };

#define WMMA_BF16(A, B, C) \
  __builtin_amdgcn_wmma_f32_16x16x32_bf16(false, (A), false, (B), (short)0, (C), false, false)

__device__ __forceinline__ v8f zero8() {
  v8f z = {0.f, 0.f, 0.f, 0.f, 0.f, 0.f, 0.f, 0.f};
  return z;
}

// One per-lane 16-byte async global->LDS copy (CDNA5).
__device__ __forceinline__ void async_copy_b128(const void* gptr, void* lptr) {
  uint32_t loff = (uint32_t)(uintptr_t)lptr;     // low 32 bits of LDS aperture addr
  uint64_t gaddr = (uint64_t)(uintptr_t)gptr;
  asm volatile("global_load_async_to_lds_b128 %0, %1, off" :: "v"(loff), "v"(gaddr) : "memory");
}
__device__ __forceinline__ void wait_async0() {
  asm volatile("s_wait_asynccnt 0x0" ::: "memory");
}

static constexpr int NB     = 8;     // batch
static constexpr int C_DIM  = 512;   // channels
static constexpr int HEADS  = 8;
static constexpr int DHEAD  = 64;
static constexpr int SEQ    = 1024;  // 32*32 pixels
static constexpr float SCALE = 0.125f;                 // 64^-0.5
static constexpr float LOG2E = 1.4426950408889634f;
static constexpr float SQRTC = 22.62741699796952f;     // sqrt(512)

// ---------------------------------------------------------------------------
// 1) Convert weights to bf16; fold SCALE*log2(e) into the Q rows of w_qkv
//    so flash attention can run its softmax in the exp2 domain.
// ---------------------------------------------------------------------------
__global__ void k_convert_weights(const float* __restrict__ wqkv,
                                  const float* __restrict__ wout,
                                  bf16_t* __restrict__ wqkv_b,
                                  bf16_t* __restrict__ wout_b) {
  int i = blockIdx.x * blockDim.x + threadIdx.x;
  const int NQ = 3 * C_DIM * C_DIM;
  const int NO = C_DIM * C_DIM;
  if (i < NQ) {
    float v = wqkv[i];
    if ((i >> 9) < C_DIM) v *= SCALE * LOG2E;   // rows [0,512) = Q channels
    wqkv_b[i] = (bf16_t)v;
  }
  if (i < NO) wout_b[i] = (bf16_t)wout[i];
}

// ---------------------------------------------------------------------------
// 2a) Per-pixel L2 norm over channels (coalesced in pixel index).
// ---------------------------------------------------------------------------
__global__ void k_rms_norms(const float* __restrict__ x, float* __restrict__ invn) {
  int p = blockIdx.x * blockDim.x + threadIdx.x;      // 0 .. 8191
  int b = p >> 10, pix = p & 1023;
  const float* xb = x + (size_t)b * C_DIM * SEQ + pix;
  float s = 0.f;
  for (int c = 0; c < C_DIM; ++c) {
    float v = xb[(size_t)c * SEQ];
    s += v * v;
  }
  float n = fmaxf(sqrtf(s), 1e-12f);
  invn[p] = SQRTC / n;
}

// ---------------------------------------------------------------------------
// 2b) Normalize + transpose to xn_t[b][p][c] bf16 via LDS tiles.
// ---------------------------------------------------------------------------
__global__ void k_norm_transpose(const float* __restrict__ x,
                                 const float* __restrict__ g,
                                 const float* __restrict__ invn,
                                 bf16_t* __restrict__ xnt) {
  __shared__ float tile[32][33];
  int tx = threadIdx.x, ty = threadIdx.y;
  int p0 = blockIdx.x * 32, c0 = blockIdx.y * 32, b = blockIdx.z;
  const float* xb = x + (size_t)b * C_DIM * SEQ;
#pragma unroll
  for (int k = 0; k < 4; ++k)
    tile[ty + 8 * k][tx] = xb[(size_t)(c0 + ty + 8 * k) * SEQ + p0 + tx];
  __syncthreads();
#pragma unroll
  for (int k = 0; k < 4; ++k) {
    int p = p0 + ty + 8 * k;
    int c = c0 + tx;
    float v = tile[tx][ty + 8 * k] * invn[b * SEQ + p] * g[c];
    xnt[((size_t)b * SEQ + p) * C_DIM + c] = (bf16_t)v;
  }
}

// ---------------------------------------------------------------------------
// 3) QKV GEMM with async-staged shared B tile.
//    Block = 8 waves = 128 out-ch x 64 pixels; wave tile 16x64.
//    B chunk (64p x 32c = 4KB) async double-buffered; A register-pipelined.
//    Fragment loads batched ahead of the 4 WMMAs.
// ---------------------------------------------------------------------------
__global__ void k_qkv_gemm(const bf16_t* __restrict__ wqkv_b,
                           const bf16_t* __restrict__ xnt,
                           bf16_t* __restrict__ qb,
                           bf16_t* __restrict__ kb,
                           bf16_t* __restrict__ vb) {
  __shared__ bf16_t ldsB[2][64][32];
  const int tid = threadIdx.x;
  const int lane = tid & 31;
  const int wib  = tid >> 5;
  int bid = blockIdx.x;
  int b  = bid / (12 * 16);
  int r  = bid % (12 * 16);
  int og = r / 16, pt = r % 16;
  int o0 = (og * 8 + wib) * 16;
  int p0 = pt * 64;
  const int half = lane >> 4;
  const int l16  = lane & 15;

  const int tp = tid >> 2, tc = (tid & 3) * 8;
  auto issueB = [&](int kc, int bufi) {
    async_copy_b128(xnt + ((size_t)b * SEQ + p0 + tp) * C_DIM + kc + tc,
                    &ldsB[bufi][tp][tc]);
  };
  issueB(0, 0);

  auto loadA = [&](int kc) {
    Frag16 a;
    const uint32_t* ap = (const uint32_t*)(wqkv_b + (size_t)(o0 + l16) * C_DIM + kc);
#pragma unroll
    for (int v = 0; v < 4; ++v) {
      a.u[v]     = ap[v + 4 * half];
      a.u[4 + v] = ap[8 + v + 4 * half];
    }
    return a;
  };

  v8f acc[4] = {zero8(), zero8(), zero8(), zero8()};
  Frag16 a_cur = loadA(0);

  for (int kc = 0; kc < C_DIM; kc += 32) {
    int bufi = (kc >> 5) & 1;
    wait_async0();
    __syncthreads();
    bool more = (kc + 32 < C_DIM);
    if (more) issueB(kc + 32, bufi ^ 1);
    Frag16 a_nxt = more ? loadA(kc + 32) : a_cur;
    Frag16 bfr[4];
#pragma unroll
    for (int n = 0; n < 4; ++n) {
      const uint32_t* bp = (const uint32_t*)&ldsB[bufi][n * 16 + l16][0];
#pragma unroll
      for (int v = 0; v < 8; ++v) bfr[n].u[v] = bp[(v & 3) + 4 * half + 8 * (v >> 2)];
    }
#pragma unroll
    for (int n = 0; n < 4; ++n) acc[n] = WMMA_BF16(a_cur.v, bfr[n].v, acc[n]);
    a_cur = a_nxt;
  }

#pragma unroll
  for (int n = 0; n < 4; ++n) {
    int p = p0 + n * 16 + l16;
#pragma unroll
    for (int rr = 0; rr < 8; ++rr) {
      int o = o0 + rr + 8 * half;
      bf16_t val = (bf16_t)acc[n][rr];
      int sect = o >> 9;         // uniform per tile
      int oc = o & 511;
      int h = oc >> 6, d = oc & 63;
      size_t bh = (size_t)b * HEADS + h;
      if (sect == 0)      qb[(bh * SEQ + p) * DHEAD + d] = val;
      else if (sect == 1) kb[(bh * SEQ + p) * DHEAD + d] = val;
      else                vb[(bh * DHEAD + d) * SEQ + p] = val;
    }
  }
}

// ---------------------------------------------------------------------------
// 4) Flash attention, 64-wide j tiles.
//    Block = 8 waves, same (b,h); wave owns 16 query rows.
//    Per tile: K 64x64 and V 64x64 (8KB each) async double-buffered;
//    S = 8 WMMAs, one exp2-domain softmax pass, P restaged via per-wave LDS,
//    PV = 8 WMMAs.
// ---------------------------------------------------------------------------
__global__ void k_flash_attn(const bf16_t* __restrict__ qb,
                             const bf16_t* __restrict__ kb,
                             const bf16_t* __restrict__ vb,
                             bf16_t* __restrict__ attO) {
  __shared__ bf16_t ldsK[2][64][64];   // [j][d]
  __shared__ bf16_t ldsV[2][64][64];   // [d][j]
  __shared__ bf16_t ldsp[8][16][66];   // per-wave 16x64 P staging (+pad)
  const int tid  = threadIdx.x;
  const int lane = tid & 31;
  const int wib  = tid >> 5;
  const int gwave = blockIdx.x * 8 + wib;
  const int bh = gwave >> 6;           // all 8 waves of a block share bh
  const int i0 = (gwave & 63) * 16;
  const int half = lane >> 4;
  const int l16  = lane & 15;

  // async staging: 64x64 bf16 tile = 8KB = 256 threads x 2 b128
  const int sr = tid >> 2, sc = (tid & 3) * 16;
  auto issueKV = [&](int jt, int bufi) {
    const bf16_t* kg = kb + ((size_t)bh * SEQ + jt + sr) * DHEAD + sc;
    async_copy_b128(kg,     &ldsK[bufi][sr][sc]);
    async_copy_b128(kg + 8, &ldsK[bufi][sr][sc + 8]);
    const bf16_t* vg = vb + ((size_t)bh * DHEAD + sr) * SEQ + jt + sc;
    async_copy_b128(vg,     &ldsV[bufi][sr][sc]);
    async_copy_b128(vg + 8, &ldsV[bufi][sr][sc + 8]);
  };
  issueKV(0, 0);

  // Q A-fragments (d chunks 0..31, 32..63); q pre-scaled by SCALE*log2(e).
  Frag16 qa[2];
  {
    const uint32_t* qp = (const uint32_t*)(qb + ((size_t)bh * SEQ + i0 + l16) * DHEAD);
#pragma unroll
    for (int c = 0; c < 2; ++c)
#pragma unroll
      for (int v = 0; v < 4; ++v) {
        qa[c].u[v]     = qp[c * 16 + v + 4 * half];
        qa[c].u[4 + v] = qp[c * 16 + 8 + v + 4 * half];
      }
  }

  float m[8], l[8];
  v8f o[4] = {zero8(), zero8(), zero8(), zero8()};
#pragma unroll
  for (int r = 0; r < 8; ++r) { m[r] = -3.0e38f; l[r] = 0.f; }

  __bf16 (*myp)[66] = ldsp[wib];

  for (int jt = 0; jt < SEQ; jt += 64) {
    int bufi = (jt >> 6) & 1;
    wait_async0();
    __syncthreads();
    if (jt + 64 < SEQ) issueKV(jt + 64, bufi ^ 1);

    // ---- S = Q * K^T (16 x 64): batch 4 K fragments, then 4 WMMAs, x2 ----
    v8f s[4] = {zero8(), zero8(), zero8(), zero8()};
#pragma unroll
    for (int c = 0; c < 2; ++c) {
      Frag16 kfr[4];
#pragma unroll
      for (int n = 0; n < 4; ++n) {
        const uint32_t* kp = (const uint32_t*)&ldsK[bufi][n * 16 + l16][c * 32];
#pragma unroll
        for (int v = 0; v < 8; ++v) kfr[n].u[v] = kp[(v & 3) + 4 * half + 8 * (v >> 2)];
      }
#pragma unroll
      for (int n = 0; n < 4; ++n) s[n] = WMMA_BF16(qa[c].v, kfr[n].v, s[n]);
    }

    // ---- online softmax (exp2 domain), one reduction pass per row ----
    float alpha[8];
#pragma unroll
    for (int r = 0; r < 8; ++r) {
      float mt = fmaxf(fmaxf(s[0][r], s[1][r]), fmaxf(s[2][r], s[3][r]));
      mt = fmaxf(mt, __shfl_xor(mt, 1, 32));
      mt = fmaxf(mt, __shfl_xor(mt, 2, 32));
      mt = fmaxf(mt, __shfl_xor(mt, 4, 32));
      mt = fmaxf(mt, __shfl_xor(mt, 8, 32));
      float nm = fmaxf(m[r], mt);
      alpha[r] = exp2f(m[r] - nm);
      m[r] = nm;
      float rs = 0.f;
#pragma unroll
      for (int t = 0; t < 4; ++t) {
        float pv = exp2f(s[t][r] - nm);
        s[t][r] = pv;
        rs += pv;
      }
      rs += __shfl_xor(rs, 1, 32);
      rs += __shfl_xor(rs, 2, 32);
      rs += __shfl_xor(rs, 4, 32);
      rs += __shfl_xor(rs, 8, 32);
      l[r] = l[r] * alpha[r] + rs;
    }
#pragma unroll
    for (int n = 0; n < 4; ++n)
#pragma unroll
      for (int r = 0; r < 8; ++r) o[n][r] *= alpha[r];

    // ---- stage P (C-layout) into per-wave LDS, reload as A-fragments ----
#pragma unroll
    for (int t = 0; t < 4; ++t)
#pragma unroll
      for (int r = 0; r < 8; ++r)
        myp[r + 8 * half][t * 16 + l16] = (bf16_t)s[t][r];
    asm volatile("s_wait_dscnt 0x0" ::: "memory");
    Frag16 pa[2];
    {
      const uint32_t* pp = (const uint32_t*)(myp[l16]);
#pragma unroll
      for (int c = 0; c < 2; ++c)
#pragma unroll
        for (int v = 0; v < 4; ++v) {
          pa[c].u[v]     = pp[c * 16 + v + 4 * half];
          pa[c].u[4 + v] = pp[c * 16 + 8 + v + 4 * half];
        }
    }

    // ---- O += P * V ----
#pragma unroll
    for (int n = 0; n < 4; ++n) {
      Frag16 vfr[2];
#pragma unroll
      for (int c = 0; c < 2; ++c) {
        const uint32_t* vp = (const uint32_t*)&ldsV[bufi][n * 16 + l16][c * 32];
#pragma unroll
        for (int v = 0; v < 8; ++v) vfr[c].u[v] = vp[(v & 3) + 4 * half + 8 * (v >> 2)];
      }
#pragma unroll
      for (int c = 0; c < 2; ++c) o[n] = WMMA_BF16(pa[c].v, vfr[c].v, o[n]);
    }
  }

  // ---- epilogue: divide by row sums, store attO[b][i][h*64+d] ----
  int b = bh >> 3, h = bh & 7;
#pragma unroll
  for (int n = 0; n < 4; ++n)
#pragma unroll
    for (int r = 0; r < 8; ++r) {
      int i = i0 + r + 8 * half;
      int d = n * 16 + l16;
      float val = o[n][r] / l[r];
      attO[((size_t)b * SEQ + i) * C_DIM + h * DHEAD + d] = (bf16_t)val;
    }
}

// ---------------------------------------------------------------------------
// 5) Out projection + bias + residual, async-staged shared B tile.
// ---------------------------------------------------------------------------
__global__ void k_out_proj(const bf16_t* __restrict__ wout_b,
                           const bf16_t* __restrict__ attO,
                           const float* __restrict__ bout,
                           const float* __restrict__ x,
                           float* __restrict__ out) {
  __shared__ bf16_t ldsB[2][64][32];
  const int tid = threadIdx.x;
  const int lane = tid & 31;
  const int wib  = tid >> 5;
  int bid = blockIdx.x;
  int b  = bid / (4 * 16);
  int r  = bid % (4 * 16);
  int og = r / 16, pt = r % 16;
  int o0 = (og * 8 + wib) * 16;
  int p0 = pt * 64;
  const int half = lane >> 4;
  const int l16  = lane & 15;

  const int tp = tid >> 2, tc = (tid & 3) * 8;
  auto issueB = [&](int kc, int bufi) {
    async_copy_b128(attO + ((size_t)b * SEQ + p0 + tp) * C_DIM + kc + tc,
                    &ldsB[bufi][tp][tc]);
  };
  issueB(0, 0);

  auto loadA = [&](int kc) {
    Frag16 a;
    const uint32_t* ap = (const uint32_t*)(wout_b + (size_t)(o0 + l16) * C_DIM + kc);
#pragma unroll
    for (int v = 0; v < 4; ++v) {
      a.u[v]     = ap[v + 4 * half];
      a.u[4 + v] = ap[8 + v + 4 * half];
    }
    return a;
  };

  v8f acc[4] = {zero8(), zero8(), zero8(), zero8()};
  Frag16 a_cur = loadA(0);

  for (int kc = 0; kc < C_DIM; kc += 32) {
    int bufi = (kc >> 5) & 1;
    wait_async0();
    __syncthreads();
    bool more = (kc + 32 < C_DIM);
    if (more) issueB(kc + 32, bufi ^ 1);
    Frag16 a_nxt = more ? loadA(kc + 32) : a_cur;
    Frag16 bfr[4];
#pragma unroll
    for (int n = 0; n < 4; ++n) {
      const uint32_t* bp = (const uint32_t*)&ldsB[bufi][n * 16 + l16][0];
#pragma unroll
      for (int v = 0; v < 8; ++v) bfr[n].u[v] = bp[(v & 3) + 4 * half + 8 * (v >> 2)];
    }
#pragma unroll
    for (int n = 0; n < 4; ++n) acc[n] = WMMA_BF16(a_cur.v, bfr[n].v, acc[n]);
    a_cur = a_nxt;
  }

#pragma unroll
  for (int n = 0; n < 4; ++n) {
    int p = p0 + n * 16 + l16;
#pragma unroll
    for (int rr = 0; rr < 8; ++rr) {
      int o = o0 + rr + 8 * half;
      size_t idx = ((size_t)b * C_DIM + o) * SEQ + p;
      out[idx] = acc[n][rr] + bout[o] + x[idx];
    }
  }
}

// ---------------------------------------------------------------------------
// Launcher
// ---------------------------------------------------------------------------
extern "C" void kernel_launch(void* const* d_in, const int* in_sizes, int n_in,
                              void* d_out, int out_size, void* d_ws, size_t ws_size,
                              hipStream_t stream) {
  const float* x     = (const float*)d_in[0];
  const float* wqkv  = (const float*)d_in[1];
  const float* wout  = (const float*)d_in[2];
  const float* bout  = (const float*)d_in[3];
  const float* g     = (const float*)d_in[4];
  float* out = (float*)d_out;

  char* ws = (char*)d_ws;
  size_t off = 0;
  bf16_t* wqkv_b = (bf16_t*)(ws + off); off += (size_t)3 * C_DIM * C_DIM * 2;
  bf16_t* wout_b = (bf16_t*)(ws + off); off += (size_t)C_DIM * C_DIM * 2;
  bf16_t* xnt    = (bf16_t*)(ws + off); off += (size_t)NB * SEQ * C_DIM * 2;
  bf16_t* qb     = (bf16_t*)(ws + off); off += (size_t)NB * HEADS * SEQ * DHEAD * 2;
  bf16_t* kb     = (bf16_t*)(ws + off); off += (size_t)NB * HEADS * SEQ * DHEAD * 2;
  bf16_t* vb     = (bf16_t*)(ws + off); off += (size_t)NB * HEADS * SEQ * DHEAD * 2;
  bf16_t* attO   = (bf16_t*)(ws + off); off += (size_t)NB * SEQ * C_DIM * 2;
  float*  invn   = (float*)(ws + off);  off += (size_t)NB * SEQ * 4;

  {
    int n = 3 * C_DIM * C_DIM;
    k_convert_weights<<<(n + 255) / 256, 256, 0, stream>>>(wqkv, wout, wqkv_b, wout_b);
  }
  k_rms_norms<<<(NB * SEQ) / 256, 256, 0, stream>>>(x, invn);
  k_norm_transpose<<<dim3(SEQ / 32, C_DIM / 32, NB), dim3(32, 8), 0, stream>>>(x, g, invn, xnt);
  // QKV gemm: 8 b * 12 o-groups * 16 p-tiles = 1536 blocks of 8 waves
  k_qkv_gemm<<<NB * 12 * 16, 256, 0, stream>>>(wqkv_b, xnt, qb, kb, vb);
  // flash attention: 64 bh * 64 i-tiles = 4096 waves -> 512 blocks
  k_flash_attn<<<(NB * HEADS * (SEQ / 16)) / 8, 256, 0, stream>>>(qb, kb, vb, attO);
  // out projection: 8 b * 4 o-groups * 16 p-tiles = 512 blocks
  k_out_proj<<<NB * 4 * 16, 256, 0, stream>>>(wout_b, attO, bout, x, out);
}
